// BouzidNet_4741643894829
// MI455X (gfx1250) — compile-verified, matrix operand order checked by reading, not run on hardware
//
#include <hip/hip_runtime.h>

typedef __attribute__((ext_vector_type(16))) _Float16 v16h;
typedef __attribute__((ext_vector_type(8)))  _Float16 v8h;
typedef __attribute__((ext_vector_type(8)))  float    v8f;

#define BS        4
#define N_ATOMS   256
#define P_RBF     128
#define F_DIM     64
#define DI        32
#define NB_INTER  3
#define GAMMA_C   10.0f
#define LOG2_C    0.69314718055994530942f
#define W1_PITCH  72   // padded LDS pitch (halfs), multiple of 8 for b128 chunks

__device__ __forceinline__ float ssp_f(float x) {
    // softplus(x) - log(2), overflow-safe
    float e = __expf(-fabsf(x));
    return fmaxf(x, 0.0f) + __logf(1.0f + e) - LOG2_C;
}

__device__ __forceinline__ v16h cat8(v8h lo, v8h hi) {
    return __builtin_shufflevector(lo, hi, 0,1,2,3,4,5,6,7,8,9,10,11,12,13,14,15);
}

// ---------------- prep: transpose+convert weights to f16 [f][k] ----------------
__global__ void k_prep_weights(const float* __restrict__ d1w, const float* __restrict__ d2w,
                               _Float16* __restrict__ d1wT, _Float16* __restrict__ d2wT) {
    int t = blockIdx.x * blockDim.x + threadIdx.x;
    const int n1 = NB_INTER * F_DIM * P_RBF;   // 24576
    const int n2 = NB_INTER * F_DIM * F_DIM;   // 12288
    if (t < n1) {
        int i = t / (F_DIM * P_RBF);
        int rem = t % (F_DIM * P_RBF);
        int f = rem / P_RBF;
        int k = rem % P_RBF;
        d1wT[t] = (_Float16)d1w[(i * P_RBF + k) * F_DIM + f];
    } else if (t < n1 + n2) {
        int u = t - n1;
        int i = u / (F_DIM * F_DIM);
        int rem = u % (F_DIM * F_DIM);
        int f = rem / F_DIM;
        int k = rem % F_DIM;
        d2wT[u] = (_Float16)d2w[(i * F_DIM + k) * F_DIM + f];
    }
}

// -------- prep: X0 = emb[Z], Xm0 = X0*mask, mask = (X0!=0), maskloss = (Z>0) ---
__global__ void k_prep_atoms(const int* __restrict__ Z, const float* __restrict__ emb,
                             float* __restrict__ X0, float* __restrict__ Xm0,
                             float* __restrict__ maskG, float* __restrict__ maskloss) {
    int b = blockIdx.x, n = threadIdx.x;
    int z = Z[b * N_ATOMS + n];
    maskloss[b * N_ATOMS + n] = (z > 0) ? 1.0f : 0.0f;
    for (int f = 0; f < F_DIM; ++f) {
        float e = emb[z * F_DIM + f];
        float mk = (e != 0.0f) ? 1.0f : 0.0f;
        size_t idx = ((size_t)b * N_ATOMS + n) * F_DIM + f;
        X0[idx] = e;
        Xm0[idx] = e * mk;
        maskG[idx] = mk;
    }
}

// ---------------- rbf materialization (f16, row-major [b][n][m][p]) ------------
__global__ void k_rbf(const float* __restrict__ R, _Float16* __restrict__ rbf) {
    int n = blockIdx.x, b = blockIdx.y, m = threadIdx.x;
    float nx = R[((size_t)b * N_ATOMS + n) * 3 + 0];
    float ny = R[((size_t)b * N_ATOMS + n) * 3 + 1];
    float nz = R[((size_t)b * N_ATOMS + n) * 3 + 2];
    float mx = R[((size_t)b * N_ATOMS + m) * 3 + 0];
    float my = R[((size_t)b * N_ATOMS + m) * 3 + 1];
    float mz = R[((size_t)b * N_ATOMS + m) * 3 + 2];
    float dx = nx - mx, dy = ny - my, dz = nz - mz;
    float d2 = dx * dx + dy * dy + dz * dz;
    float dist = (d2 > 0.0f) ? sqrtf(d2) : 0.0f;
    _Float16* out = rbf + (((size_t)(b * N_ATOMS + n)) * N_ATOMS + m) * P_RBF;
    const float step = 30.0f / 127.0f;
    for (int p0 = 0; p0 < P_RBF; p0 += 8) {
        v8h v;
        #pragma unroll
        for (int j = 0; j < 8; ++j) {
            float d = step * (float)(p0 + j) - dist;
            v[j] = (_Float16)__expf(-GAMMA_C * d * d);
        }
        *(v8h*)(out + p0) = v;
    }
}

// ---------------- fused interaction block (one workgroup per (b,n)) ------------
__global__ void __launch_bounds__(256)
k_interact(const _Float16* __restrict__ rbf,
           const _Float16* __restrict__ d1wT,  // [64][128] f16 (transposed)
           const _Float16* __restrict__ d2wT,  // [64][64]  f16 (transposed)
           const float* __restrict__ d1b, const float* __restrict__ d2b,
           const float* __restrict__ w2, const float* __restrict__ b2,
           const float* __restrict__ w3, const float* __restrict__ b3,
           const float* __restrict__ Xin, const float* __restrict__ XmIn,
           const float* __restrict__ maskG,
           float* __restrict__ Xout, float* __restrict__ XmOut) {
    __shared__ _Float16 w1[N_ATOMS * W1_PITCH];  // 36 KB: GEMM1 out / GEMM2 in
    __shared__ float convS[F_DIM];
    __shared__ float cmS[F_DIM];
    __shared__ float r1S[F_DIM];
    __shared__ float partS[4][F_DIM];            // K-split partials for matvecs

    const int tid  = threadIdx.x;
    const int lane = tid & 31;
    const int wave = tid >> 5;
    const int lq   = lane & 15;   // row/col within tile
    const int hs   = lane >> 4;   // K half-select
    const int bn   = blockIdx.x;
    const int b    = bn >> 8;
    const int n    = bn & 255;

    const _Float16* rbf_bn = rbf + (size_t)bn * N_ATOMS * P_RBF;

    // Warm caches for this block's 64KB rbf slice (global_prefetch_b8).
    __builtin_prefetch(rbf_bn + (size_t)tid * 128, 0, 3);

    if (tid < F_DIM) convS[tid] = 0.0f;
    __syncthreads();

    // ============ GEMM1: W1 = ssp(rbf @ d1w + d1b), kept in LDS as f16 ========
    #pragma unroll
    for (int mi = 0; mi < 2; ++mi) {
        const int mt = wave * 2 + mi;
        v16h afr[4];
        #pragma unroll
        for (int kt = 0; kt < 4; ++kt) {
            const _Float16* p = rbf_bn + (size_t)(mt * 16 + lq) * P_RBF + kt * 32 + hs * 8;
            afr[kt] = cat8(*(const v8h*)p, *(const v8h*)(p + 16));
        }
        #pragma unroll
        for (int ft = 0; ft < 4; ++ft) {
            v8f acc = {};
            #pragma unroll
            for (int kt = 0; kt < 4; ++kt) {
                const _Float16* q = d1wT + (size_t)(ft * 16 + lq) * P_RBF + kt * 32 + hs * 8;
                v16h bfr = cat8(*(const v8h*)q, *(const v8h*)(q + 16));
                acc = __builtin_amdgcn_wmma_f32_16x16x32_f16(
                        false, afr[kt], false, bfr, (short)0, acc, false, false);
            }
            const int f = ft * 16 + lq;
            const float bias = d1b[f];
            #pragma unroll
            for (int r = 0; r < 8; ++r) {
                const int m = mt * 16 + hs * 8 + r;   // D layout: M = vgpr + 8*hs
                w1[m * W1_PITCH + f] = (_Float16)ssp_f(acc[r] + bias);
            }
        }
    }
    // w1 produced and consumed by the same wave (same m-tiles) -> no barrier.

    // ============ GEMM2: W2 = ssp(W1 @ d2w + d2b); conv += W2 * Xm ============
    #pragma unroll
    for (int mi = 0; mi < 2; ++mi) {
        const int mt = wave * 2 + mi;
        v16h afr[2];
        #pragma unroll
        for (int kt = 0; kt < 2; ++kt) {
            const _Float16* p = &w1[(mt * 16 + lq) * W1_PITCH + kt * 32 + hs * 8];
            afr[kt] = cat8(*(const v8h*)p, *(const v8h*)(p + 16));
        }
        #pragma unroll
        for (int ft = 0; ft < 4; ++ft) {
            v8f acc = {};
            #pragma unroll
            for (int kt = 0; kt < 2; ++kt) {
                const _Float16* q = d2wT + (size_t)(ft * 16 + lq) * F_DIM + kt * 32 + hs * 8;
                v16h bfr = cat8(*(const v8h*)q, *(const v8h*)(q + 16));
                acc = __builtin_amdgcn_wmma_f32_16x16x32_f16(
                        false, afr[kt], false, bfr, (short)0, acc, false, false);
            }
            const int f = ft * 16 + lq;
            const float bias = d2b[f];
            float part = 0.0f;
            #pragma unroll
            for (int r = 0; r < 8; ++r) {
                const int m = mt * 16 + hs * 8 + r;
                float wv = ssp_f(acc[r] + bias);
                part += wv * XmIn[((size_t)b * N_ATOMS + m) * F_DIM + f];
            }
            atomicAdd(&convS[f], part);               // ds_add_f32
        }
    }
    __syncthreads();

    // ============ tiny matvec chain (K split over 4 lane groups) ==============
    if (tid < F_DIM)
        cmS[tid] = convS[tid] * maskG[((size_t)b * N_ATOMS + n) * F_DIM + tid];
    __syncthreads();
    {
        const int f = tid & 63, c = tid >> 6;
        float s = 0.0f;
        #pragma unroll
        for (int k = c * 16; k < c * 16 + 16; ++k) s += cmS[k] * w2[k * F_DIM + f];
        partS[c][f] = s;
    }
    __syncthreads();
    if (tid < F_DIM) {
        float s = b2[tid] + partS[0][tid] + partS[1][tid] + partS[2][tid] + partS[3][tid];
        s *= maskG[((size_t)b * N_ATOMS + n) * F_DIM + tid];
        r1S[tid] = ssp_f(s);
    }
    __syncthreads();
    {
        const int f = tid & 63, c = tid >> 6;
        float s = 0.0f;
        #pragma unroll
        for (int k = c * 16; k < c * 16 + 16; ++k) s += r1S[k] * w3[k * F_DIM + f];
        partS[c][f] = s;
    }
    __syncthreads();
    if (tid < F_DIM) {
        size_t idx = ((size_t)b * N_ATOMS + n) * F_DIM + tid;
        float mk = maskG[idx];
        float s = (b3[tid] + partS[0][tid] + partS[1][tid] + partS[2][tid] + partS[3][tid]) * mk;
        float xnew = Xin[idx] + s;
        Xout[idx] = xnew;
        XmOut[idx] = xnew * mk;
    }
}

// ---------------- final head + per-batch reduction -----------------------------
__global__ void k_final(const float* __restrict__ X, const float* __restrict__ maskG,
                        const float* __restrict__ maskloss,
                        const float* __restrict__ wi, const float* __restrict__ bi,
                        const float* __restrict__ wf, const float* __restrict__ bf,
                        float* __restrict__ out) {
    __shared__ float red[N_ATOMS];
    int b = blockIdx.x, n = threadIdx.x;
    const float* xr = X + ((size_t)b * N_ATOMS + n) * F_DIM;
    const float* mr = maskG + ((size_t)b * N_ATOMS + n) * F_DIM;
    float o_acc = 0.0f;
    for (int d = 0; d < DI; ++d) {
        float s = bi[d];
        for (int f = 0; f < F_DIM; ++f) s += xr[f] * wi[f * DI + d];
        s *= mr[d];                       // mask[..., :DI] before ssp
        o_acc += ssp_f(s) * wf[d];
    }
    float o = (o_acc + bf[0]) * maskloss[b * N_ATOMS + n];
    red[n] = o;
    __syncthreads();
    for (int s = 128; s > 0; s >>= 1) {
        if (n < s) red[n] += red[n + s];
        __syncthreads();
    }
    if (n == 0) out[b] = red[0];
}

extern "C" void kernel_launch(void* const* d_in, const int* in_sizes, int n_in,
                              void* d_out, int out_size, void* d_ws, size_t ws_size,
                              hipStream_t stream) {
    const int*   Z   = (const int*)  d_in[0];
    const float* R   = (const float*)d_in[1];
    const float* emb = (const float*)d_in[2];
    const float* d1w = (const float*)d_in[3];
    const float* d1b = (const float*)d_in[4];
    const float* d2w = (const float*)d_in[5];
    const float* d2b = (const float*)d_in[6];
    const float* w2  = (const float*)d_in[7];
    const float* b2  = (const float*)d_in[8];
    const float* w3  = (const float*)d_in[9];
    const float* b3  = (const float*)d_in[10];
    const float* wi  = (const float*)d_in[11];
    const float* bi  = (const float*)d_in[12];
    const float* wf  = (const float*)d_in[13];
    const float* bf  = (const float*)d_in[14];
    float* out = (float*)d_out;

    char* ws = (char*)d_ws;
    size_t off = 0;
    _Float16* rbf  = (_Float16*)(ws + off); off += (size_t)BS * N_ATOMS * N_ATOMS * P_RBF * 2;
    _Float16* d1wT = (_Float16*)(ws + off); off += (size_t)NB_INTER * F_DIM * P_RBF * 2;
    _Float16* d2wT = (_Float16*)(ws + off); off += (size_t)NB_INTER * F_DIM * F_DIM * 2;
    float* Xa       = (float*)(ws + off);   off += (size_t)BS * N_ATOMS * F_DIM * 4;
    float* Xb       = (float*)(ws + off);   off += (size_t)BS * N_ATOMS * F_DIM * 4;
    float* Xma      = (float*)(ws + off);   off += (size_t)BS * N_ATOMS * F_DIM * 4;
    float* Xmb      = (float*)(ws + off);   off += (size_t)BS * N_ATOMS * F_DIM * 4;
    float* maskG    = (float*)(ws + off);   off += (size_t)BS * N_ATOMS * F_DIM * 4;
    float* maskloss = (float*)(ws + off);   off += (size_t)BS * N_ATOMS * 4;

    k_prep_weights<<<144, 256, 0, stream>>>(d1w, d2w, d1wT, d2wT);
    k_prep_atoms<<<BS, 256, 0, stream>>>(Z, emb, Xa, Xma, maskG, maskloss);
    k_rbf<<<dim3(N_ATOMS, BS), 256, 0, stream>>>(R, rbf);

    const float* xin = Xa;  const float* xmin = Xma;
    float* xout = Xb;       float* xmout = Xmb;
    for (int i = 0; i < NB_INTER; ++i) {
        k_interact<<<BS * N_ATOMS, 256, 0, stream>>>(
            rbf,
            d1wT + (size_t)i * F_DIM * P_RBF,
            d2wT + (size_t)i * F_DIM * F_DIM,
            d1b + i * F_DIM, d2b + i * F_DIM,
            w2 + (size_t)i * F_DIM * F_DIM, b2 + i * F_DIM,
            w3 + (size_t)i * F_DIM * F_DIM, b3 + i * F_DIM,
            xin, xmin, maskG, xout, xmout);
        const float* t1 = xout;  xout  = (float*)xin;  xin  = t1;
        const float* t2 = xmout; xmout = (float*)xmin; xmin = t2;
    }
    k_final<<<BS, 256, 0, stream>>>(xin, maskG, maskloss, wi, bi, wf, bf, out);
}
